// ArxivReco_48464410968124
// MI455X (gfx1250) — matrix operand order repaired and checked by reading, not compiled, not run on hardware
//
#include <hip/hip_runtime.h>

typedef __attribute__((ext_vector_type(16))) _Float16 v16h;
typedef __attribute__((ext_vector_type(8)))  _Float16 v8h;
typedef __attribute__((ext_vector_type(8)))  float    v8f;

static __device__ __forceinline__ void atomic_add_f32(float* p, float v) {
  __hip_atomic_fetch_add(p, v, __ATOMIC_RELAXED, __HIP_MEMORY_SCOPE_AGENT);
}

// ---------------- degree / normalization ----------------

__global__ void init_deg_kernel(float* __restrict__ deg, int n) {
  int i = blockIdx.x * blockDim.x + threadIdx.x;
  if (i < n) deg[i] = 1.0f;  // self-loop contributes 1
}

__global__ void edge_deg_kernel(const int* __restrict__ dst, float* __restrict__ deg, int E) {
  int e = blockIdx.x * blockDim.x + threadIdx.x;
  if (e < E) atomic_add_f32(&deg[dst[e]], 1.0f);
}

__global__ void rsqrt_kernel(float* __restrict__ deg, int n) {
  int i = blockIdx.x * blockDim.x + threadIdx.x;
  if (i < n) {
    float d = deg[i];
    deg[i] = (d > 0.0f) ? rsqrtf(d) : 0.0f;
  }
}

__global__ void edge_norm_kernel(const int* __restrict__ src, const int* __restrict__ dst,
                                 const float* __restrict__ dinv, float* __restrict__ enorm, int E) {
  int e = blockIdx.x * blockDim.x + threadIdx.x;
  if (e < E) enorm[e] = dinv[src[e]] * dinv[dst[e]];
}

// ---------------- f32 -> f16 packing ----------------

__global__ void f32_to_f16_kernel(const float* __restrict__ src, _Float16* __restrict__ dst,
                                  long long n) {
  long long i = (long long)blockIdx.x * blockDim.x + threadIdx.x;
  if (i < n) dst[i] = (_Float16)src[i];
}

// W[K][NOUT] f32 (row-major) -> Wt[NOUT][K] f16 (each output column contiguous over K)
__global__ void wt_convert_kernel(const float* __restrict__ W, _Float16* __restrict__ Wt,
                                  int K, int NOUT) {
  int i = blockIdx.x * blockDim.x + threadIdx.x;
  if (i < K * NOUT) {
    int k = i / NOUT, n = i % NOUT;
    Wt[(size_t)n * K + k] = (_Float16)W[i];
  }
}

// ---------------- WMMA GEMM:  H[N,NOUT] = A[N,K](f16) @ W  via Bt[NOUT,K](f16) ----------------
// One wave computes a 16 x (16*NT) macro-tile; A fragment reused across NT WMMAs.

template <int K, int NOUT, int NT>
__global__ void gemm_wmma_f16_kernel(const _Float16* __restrict__ A,
                                     const _Float16* __restrict__ Bt,
                                     float* __restrict__ H, int nrows) {
  const int lane = threadIdx.x & 31;           // wave32
  const int wave = threadIdx.x >> 5;
  const int gw   = blockIdx.x * (blockDim.x >> 5) + wave;
  constexpr int colMacros = NOUT / (16 * NT);
  const int rowTiles = nrows / 16;             // N divisible by 16 (N=100000)
  if (gw >= rowTiles * colMacros) return;      // wave-uniform guard (EXEC all-ones)

  const int m0   = (gw / colMacros) * 16;
  const int n0   = (gw % colMacros) * (16 * NT);
  const int half = lane >> 4;                  // 0: lanes 0-15, 1: lanes 16-31
  const int l15  = lane & 15;

  const _Float16* arow = A + (size_t)(m0 + l15) * K;
  v8f c[NT];
#pragma unroll
  for (int t = 0; t < NT; ++t) c[t] = (v8f){};

#pragma unroll
  for (int kk = 0; kk < K; kk += 32) {
    const int kba = kk + half * 16;            // A: half0 -> K 0..15; half1 -> K 16..31
    const int kbb = kk + half * 16;            // B: half0 -> K 0..15; half1 -> K 16..31
    v8h alo = *(const v8h*)(arow + kba);       // 16B contiguous
    v8h ahi = *(const v8h*)(arow + kba + 8);
    v16h a = __builtin_shufflevector(alo, ahi,
             0, 1, 2, 3, 4, 5, 6, 7, 8, 9, 10, 11, 12, 13, 14, 15);
#pragma unroll
    for (int t = 0; t < NT; ++t) {
      const _Float16* bp = Bt + (size_t)(n0 + t * 16 + l15) * K + kbb;
      v8h blo = *(const v8h*)(bp);             // 32B contiguous per lane
      v8h bhi = *(const v8h*)(bp + 8);
      v16h b = __builtin_shufflevector(blo, bhi,
               0, 1, 2, 3, 4, 5, 6, 7, 8, 9, 10, 11, 12, 13, 14, 15);
      c[t] = __builtin_amdgcn_wmma_f32_16x16x32_f16(false, a, false, b,
                                                    (short)0, c[t], false, false);
    }
  }

  const int orow = m0 + half * 8;              // D: VGPR r -> row r (half0) / row 8+r (half1)
#pragma unroll
  for (int t = 0; t < NT; ++t) {
    const int col = n0 + t * 16 + l15;
#pragma unroll
    for (int r = 0; r < 8; ++r)
      H[(size_t)(orow + r) * NOUT + col] = c[t][r];
  }
}

// ---------------- zero (float4) ----------------

__global__ void zero4_kernel(float4* __restrict__ p, long long n4) {
  long long i = (long long)blockIdx.x * blockDim.x + threadIdx.x;
  if (i < n4) p[i] = make_float4(0.f, 0.f, 0.f, 0.f);
}

// ---------------- edge scatter:  agg[dst] += h[src] * enorm[e] ----------------

template <int F>
__global__ void scatter_kernel(const float* __restrict__ h,
                               const int* __restrict__ src,
                               const int* __restrict__ dst,
                               const float* __restrict__ enorm,
                               float* __restrict__ agg, int E) {
  constexpr int F4 = F / 4;
  long long idx = (long long)blockIdx.x * blockDim.x + threadIdx.x;
  if (idx >= (long long)E * F4) return;
  int e = (int)(idx / F4);
  int f = (int)(idx % F4) * 4;
  int s = src[e], d = dst[e];
  float w = enorm[e];
  float4 hv = *(const float4*)(h + (size_t)s * F + f);
  float* out = agg + (size_t)d * F + f;
  atomic_add_f32(out + 0, hv.x * w);
  atomic_add_f32(out + 1, hv.y * w);
  atomic_add_f32(out + 2, hv.z * w);
  atomic_add_f32(out + 3, hv.w * w);
}

// ---------------- self-loop + bias + relu (in place on agg; fused f16 pack) ----------------

template <int F, bool WRITE_F16>
__global__ void bias_relu_kernel(float* __restrict__ agg,
                                 const float* __restrict__ h,
                                 const float* __restrict__ dinv,
                                 const float* __restrict__ bias,
                                 _Float16* __restrict__ outh, int n) {
  long long idx = (long long)blockIdx.x * blockDim.x + threadIdx.x;
  if (idx >= (long long)n * F) return;
  int i = (int)(idx / F);
  int f = (int)(idx % F);
  float di = dinv[i];
  float v  = agg[idx] + h[idx] * di * di + bias[f];
  v = v > 0.0f ? v : 0.0f;
  agg[idx] = v;
  if (WRITE_F16) outh[idx] = (_Float16)v;
}

// ---------------- edge classifier: out[e] = [h3[s], h3[d]] . Wc + bc ----------------

__global__ void edge_out_kernel(const float* __restrict__ h3,
                                const int* __restrict__ src,
                                const int* __restrict__ dst,
                                const float* __restrict__ Wc,
                                const float* __restrict__ bc,
                                float* __restrict__ out, int E) {
  int e = blockIdx.x * blockDim.x + threadIdx.x;
  if (e >= E) return;
  int s = src[e], d = dst[e];
  const float4* hs = (const float4*)(h3 + (size_t)s * 32);
  const float4* hd = (const float4*)(h3 + (size_t)d * 32);
  const float4* ws = (const float4*)Wc;
  const float4* wd = (const float4*)(Wc + 32);
  float acc = bc[0];
#pragma unroll
  for (int j = 0; j < 8; ++j) {
    float4 a = hs[j], b = ws[j];
    acc += a.x * b.x + a.y * b.y + a.z * b.z + a.w * b.w;
    float4 p = hd[j], q = wd[j];
    acc += p.x * q.x + p.y * q.y + p.z * q.z + p.w * q.w;
  }
  out[e] = acc;
}

// ---------------- launch ----------------

extern "C" void kernel_launch(void* const* d_in, const int* in_sizes, int n_in,
                              void* d_out, int out_size, void* d_ws, size_t ws_size,
                              hipStream_t stream) {
  const float* x  = (const float*)d_in[0];
  const int*   ei = (const int*)d_in[1];
  const float* W1 = (const float*)d_in[2];
  const float* b1 = (const float*)d_in[3];
  const float* W2 = (const float*)d_in[4];
  const float* b2 = (const float*)d_in[5];
  const float* W3 = (const float*)d_in[6];
  const float* b3 = (const float*)d_in[7];
  const float* Wc = (const float*)d_in[8];
  const float* bc = (const float*)d_in[9];
  float* out = (float*)d_out;

  const int N = in_sizes[0] / 128;
  const int E = in_sizes[1] / 2;
  const int* src = ei;
  const int* dst = ei + E;

  // workspace layout (all offsets multiples of 16 bytes)
  float*     dinv  = (float*)d_ws;                    // N f32
  float*     enorm = dinv + N;                        // E f32
  float*     h     = enorm + E;                       // N*128 f32 (GEMM out / scatter in)
  float*     agg   = h + (size_t)N * 128;             // N*128 f32 (scatter out, relu in-place)
  _Float16*  xf16  = (_Float16*)(agg + (size_t)N * 128); // N*128 f16 (GEMM A input)
  _Float16*  wt16  = xf16 + (size_t)N * 128;          // 128*128 f16 (transposed weights)

  const int B = 256;
  const int wpb = B / 32;  // waves per block

  // degrees -> dinv -> per-edge norm
  init_deg_kernel<<<(N + B - 1) / B, B, 0, stream>>>(dinv, N);
  edge_deg_kernel<<<(E + B - 1) / B, B, 0, stream>>>(dst, dinv, E);
  rsqrt_kernel<<<(N + B - 1) / B, B, 0, stream>>>(dinv, N);
  edge_norm_kernel<<<(E + B - 1) / B, B, 0, stream>>>(src, dst, dinv, enorm, E);

  // pack x to f16 once
  {
    long long n = (long long)N * 128;
    f32_to_f16_kernel<<<(int)((n + B - 1) / B), B, 0, stream>>>(x, xf16, n);
  }

  // ---- layer 1: 128 -> 128 (macro-tile 16x64, NT=4) ----
  {
    wt_convert_kernel<<<(128 * 128 + B - 1) / B, B, 0, stream>>>(W1, wt16, 128, 128);
    int waves = (N / 16) * (128 / 64);
    gemm_wmma_f16_kernel<128, 128, 4><<<(waves + wpb - 1) / wpb, B, 0, stream>>>(xf16, wt16, h, N);
    long long nz = (long long)N * 128;
    zero4_kernel<<<(int)((nz / 4 + B - 1) / B), B, 0, stream>>>((float4*)agg, nz / 4);
    long long work = (long long)E * (128 / 4);
    scatter_kernel<128><<<(int)((work + B - 1) / B), B, 0, stream>>>(h, src, dst, enorm, agg, E);
    bias_relu_kernel<128, true><<<(int)((nz + B - 1) / B), B, 0, stream>>>(agg, h, dinv, b1, xf16, N);
  }

  // ---- layer 2: 128 -> 64 (macro-tile 16x64, NT=4) ----
  {
    wt_convert_kernel<<<(128 * 64 + B - 1) / B, B, 0, stream>>>(W2, wt16, 128, 64);
    int waves = (N / 16) * (64 / 64);
    gemm_wmma_f16_kernel<128, 64, 4><<<(waves + wpb - 1) / wpb, B, 0, stream>>>(xf16, wt16, h, N);
    long long nz = (long long)N * 64;
    zero4_kernel<<<(int)((nz / 4 + B - 1) / B), B, 0, stream>>>((float4*)agg, nz / 4);
    long long work = (long long)E * (64 / 4);
    scatter_kernel<64><<<(int)((work + B - 1) / B), B, 0, stream>>>(h, src, dst, enorm, agg, E);
    bias_relu_kernel<64, true><<<(int)((nz + B - 1) / B), B, 0, stream>>>(agg, h, dinv, b2, xf16, N);
  }

  // ---- layer 3: 64 -> 32 (macro-tile 16x32, NT=2) ----
  {
    wt_convert_kernel<<<(64 * 32 + B - 1) / B, B, 0, stream>>>(W3, wt16, 64, 32);
    int waves = (N / 16) * (32 / 32);
    gemm_wmma_f16_kernel<64, 32, 2><<<(waves + wpb - 1) / wpb, B, 0, stream>>>(xf16, wt16, h, N);
    long long nz = (long long)N * 32;
    zero4_kernel<<<(int)((nz / 4 + B - 1) / B), B, 0, stream>>>((float4*)agg, nz / 4);
    long long work = (long long)E * (32 / 4);
    scatter_kernel<32><<<(int)((work + B - 1) / B), B, 0, stream>>>(h, src, dst, enorm, agg, E);
    bias_relu_kernel<32, false><<<(int)((nz + B - 1) / B), B, 0, stream>>>(agg, h, dinv, b3, (_Float16*)nullptr, N);
  }

  // ---- edge classifier ----
  edge_out_kernel<<<(E + B - 1) / B, B, 0, stream>>>(agg, src, dst, Wc, bc, out, E);
}
